// SelfAttentionEncoder_17592186044763
// MI455X (gfx1250) — matrix operand order repaired and checked by reading, not compiled, not run on hardware
//
#include <hip/hip_runtime.h>
#include <hip/hip_bf16.h>

typedef __attribute__((ext_vector_type(16))) __bf16         v16bf;
typedef __attribute__((ext_vector_type(8)))  float          v8f;
typedef __attribute__((ext_vector_type(16))) unsigned short u16x16;
typedef __attribute__((ext_vector_type(8)))  unsigned short u16x8;
typedef __attribute__((ext_vector_type(4)))  unsigned short u16x4;
typedef __attribute__((ext_vector_type(4)))  float          f4;

__device__ __forceinline__ unsigned short f2bf(float f) {
    unsigned int u = __builtin_bit_cast(unsigned int, f);
    u = (u + 0x7FFFu + ((u >> 16) & 1u)) >> 16;   // round-to-nearest-even
    return (unsigned short)u;
}

// A-fragment (16x32 bf16): per-lane K chunks p[0..7] and p[16..23]
// (caller pre-offsets p by +8*hi)
__device__ __forceinline__ v16bf a_frag_from_bf16(const unsigned short* p) {
    u16x8 lo = *(const u16x8*)p;
    u16x8 hh = *(const u16x8*)(p + 16);
    u16x16 u;
#pragma unroll
    for (int i = 0; i < 8; ++i) { u[i] = lo[i]; u[8 + i] = hh[i]; }
    return __builtin_bit_cast(v16bf, u);
}

__device__ __forceinline__ v16bf a_frag_from_f32(const float* p) {
    f4 c0 = ((const f4*)p)[0];
    f4 c1 = ((const f4*)p)[1];
    f4 c2 = ((const f4*)(p + 16))[0];
    f4 c3 = ((const f4*)(p + 16))[1];
    u16x16 u;
#pragma unroll
    for (int i = 0; i < 4; ++i) {
        u[i]      = f2bf(c0[i]);
        u[4 + i]  = f2bf(c1[i]);
        u[8 + i]  = f2bf(c2[i]);
        u[12 + i] = f2bf(c3[i]);
    }
    return __builtin_bit_cast(v16bf, u);
}

// B-fragment (32x16 bf16): per-lane K chunks p[0..7] and p[16..23] (caller pre-offsets +8*hi)
__device__ __forceinline__ v16bf b_frag_from_bf16(const unsigned short* p) {
    return a_frag_from_bf16(p);
}

// ---------------------------------------------------------------------------
// Prep: bf16 copies of X and the three weight matrices.
// ---------------------------------------------------------------------------
__global__ __launch_bounds__(256) void cvt_x_bf16(const float* __restrict__ X,
                                                  unsigned short* __restrict__ Xb) {
    const size_t i = ((size_t)blockIdx.x * 256 + threadIdx.x) * 4;
    f4 v = *(const f4*)(X + i);
    u16x4 o;
#pragma unroll
    for (int j = 0; j < 4; ++j) o[j] = f2bf(v[j]);
    *(u16x4*)(Xb + i) = o;
}

__global__ __launch_bounds__(256) void cvt_w_bf16(const float* __restrict__ Wq,
                                                  const float* __restrict__ Wk,
                                                  const float* __restrict__ Wv,
                                                  unsigned short* __restrict__ Wb) {
    const int i = blockIdx.x * 256 + threadIdx.x;    // 65536 per matrix
    Wb[i]           = f2bf(Wq[i]);
    Wb[65536 + i]   = f2bf(Wk[i]);
    Wb[131072 + i]  = f2bf(Wv[i]);
}

// ---------------------------------------------------------------------------
// Kernel 1: QKV projection from bf16 X/W.  Each wave: 16x64 output
// (4 accumulators sharing one A-fragment per K-step).
// blockIdx.y: 0 -> Q (f32 residual stream), 1 -> K (bf16 [b,n,256]),
//             2 -> V (bf16 transposed [b,h,32,512]).
// ---------------------------------------------------------------------------
__global__ __launch_bounds__(256) void qkv_wmma(
    const unsigned short* __restrict__ Xb, const unsigned short* __restrict__ Wb,
    const float* __restrict__ bq, const float* __restrict__ bk,
    const float* __restrict__ bv,
    float* __restrict__ Qout, unsigned short* __restrict__ Kout,
    unsigned short* __restrict__ VtOut)
{
    const int wave = threadIdx.x >> 5;
    const int lane = threadIdx.x & 31;
    const int ln = lane & 15, hi = lane >> 4;
    const int job = blockIdx.x * 8 + wave;      // 4096 jobs
    const int mt  = job >> 2;                   // row-tile   [0,1024)
    const int ntg = job & 3;                    // 64-col group
    const int mode = blockIdx.y;

    const unsigned short* W = Wb + (size_t)mode * 65536;
    const float* bias = (mode == 0) ? bq : (mode == 1 ? bk : bv);

    const unsigned short* ap = Xb + (size_t)(mt * 16 + ln) * 256 + 8 * hi;

    v8f acc[4] = {{}, {}, {}, {}};
#pragma unroll
    for (int kk = 0; kk < 8; ++kk) {            // K = 256 = 8 * 32
        v16bf a = a_frag_from_bf16(ap + kk * 32);
#pragma unroll
        for (int t = 0; t < 4; ++t) {
            const unsigned short* bp =
                W + (size_t)(ntg * 64 + t * 16 + ln) * 256 + 8 * hi + kk * 32;
            v16bf b = b_frag_from_bf16(bp);
            acc[t] = __builtin_amdgcn_wmma_f32_16x16x32_bf16(
                         false, a, false, b, (short)0, acc[t], false, false);
        }
    }

#pragma unroll
    for (int t = 0; t < 4; ++t) {
        const int col = ntg * 64 + t * 16 + ln;
        const float bb = bias[col];
        if (mode == 0) {
#pragma unroll
            for (int v = 0; v < 8; ++v)
                Qout[(size_t)(mt * 16 + v + 8 * hi) * 256 + col] = acc[t][v] + bb;
        } else if (mode == 1) {
#pragma unroll
            for (int v = 0; v < 8; ++v)
                Kout[(size_t)(mt * 16 + v + 8 * hi) * 256 + col] =
                    f2bf(acc[t][v] + bb);
        } else {
            const int h = col >> 5, d = col & 31;
            const int bidx  = (mt * 16) >> 9;
            const int mbase = (mt * 16) & 511;
            unsigned short* vt = VtOut + ((size_t)(bidx * 8 + h) * 32 + d) * 512;
#pragma unroll
            for (int v = 0; v < 8; ++v)
                vt[mbase + v + 8 * hi] = f2bf(acc[t][v] + bb);
        }
    }
}

// ---------------------------------------------------------------------------
// Kernel 2: one attention layer, y = x + MHA(x, K, V).
// 64 threads = 2 independent waves; wave owns 16 query rows of one (b,h).
// ---------------------------------------------------------------------------
#define SSTR 516   // f32 score-row stride (516 % 8 != 0 -> no 8-row bank alias)
#define PSTR 520   // bf16 prob-row stride (mult. of 8 -> 16B-aligned chunks)

__global__ __launch_bounds__(64) void attn_layer(
    const float* __restrict__ x, const unsigned short* __restrict__ Kb,
    const unsigned short* __restrict__ Vt, float* __restrict__ y)
{
    extern __shared__ float lds[];
    const int wave = threadIdx.x >> 5;
    const int lane = threadIdx.x & 31;
    const int ln = lane & 15, hi = lane >> 4;

    float* S = lds + wave * (16 * SSTR);                        // f32 scores
    unsigned short* Pall = (unsigned short*)(lds + 2 * (16 * SSTR));
    unsigned short* P = Pall + wave * (16 * PSTR);              // bf16 probs
    float* rs = (float*)(Pall + 2 * (16 * PSTR)) + wave * 16;   // 1/sum

    const int blk = blockIdx.x;
    const int qb = blk & 15;
    const int h  = (blk >> 4) & 7;
    const int b  = blk >> 7;
    const int qbase = qb * 32 + wave * 16;

    // Q fragment: head_dim = 32 = exactly one WMMA K-step, loaded once.
    const float* qp = x + ((size_t)(b * 512 + qbase + ln)) * 256 + h * 32 + 8 * hi;
    const v16bf qa = a_frag_from_f32(qp);

    // ---- Phase 1: S = (Q K^T) * scale, staged in LDS --------------------
    const unsigned short* kp = Kb + ((size_t)(b * 512)) * 256 + h * 32 + 8 * hi;
    const float scale = 0.1767766952966369f;    // 1/sqrt(32)
    for (int mtile = 0; mtile < 32; ++mtile) {
        v16bf kf = b_frag_from_bf16(kp + (size_t)(mtile * 16 + ln) * 256);
        __builtin_prefetch(kp + (size_t)((((mtile + 1) & 31) * 16) + ln) * 256, 0, 1);
        v8f s = {};
        s = __builtin_amdgcn_wmma_f32_16x16x32_bf16(
                false, qa, false, kf, (short)0, s, false, false);
#pragma unroll
        for (int v = 0; v < 8; ++v)
            S[(v + 8 * hi) * SSTR + mtile * 16 + ln] = s[v] * scale;
    }
    __syncthreads();

    // ---- Phase 2: softmax; bf16 probs to LDS; 1/sum deferred ------------
    {
        const int r = lane >> 1, half = lane & 1;   // 2 lanes per row
        const float* row = S + r * SSTR + half * 256;
        unsigned int* prow = (unsigned int*)(P + r * PSTR + half * 256);
        float mx = -3.0e38f;
        for (int i = 0; i < 256; ++i) mx = fmaxf(mx, row[i]);
        mx = fmaxf(mx, __shfl_xor(mx, 1, 32));
        float sum = 0.f;
        for (int i = 0; i < 256; i += 2) {
            float e0 = __expf(row[i] - mx);
            float e1 = __expf(row[i + 1] - mx);
            sum += e0 + e1;
            prow[i >> 1] = (unsigned int)f2bf(e0) | ((unsigned int)f2bf(e1) << 16);
        }
        sum += __shfl_xor(sum, 1, 32);
        if (half == 0) rs[r] = 1.0f / sum;
    }
    __syncthreads();

    // ---- Phase 3: O = P @ V, fused residual + 1/sum scaling -------------
    const unsigned short* Vb = Vt + ((size_t)(b * 8 + h) * 32) * 512;
#pragma unroll
    for (int dt = 0; dt < 2; ++dt) {            // two 16-wide head-dim tiles
        v8f acc = {};
#pragma unroll 4
        for (int mm = 0; mm < 16; ++mm) {       // K = 512 = 16 * 32
            v16bf a  = a_frag_from_bf16(P + ln * PSTR + mm * 32 + 8 * hi);
            v16bf bf = b_frag_from_bf16(Vb + (size_t)(dt * 16 + ln) * 512
                                           + mm * 32 + 8 * hi);
            acc = __builtin_amdgcn_wmma_f32_16x16x32_bf16(
                      false, a, false, bf, (short)0, acc, false, false);
        }
#pragma unroll
        for (int v = 0; v < 8; ++v) {
            const int ri = v + 8 * hi;
            const float rc = rs[ri];
            const size_t o = ((size_t)(b * 512 + qbase + ri)) * 256
                             + h * 32 + dt * 16 + ln;
            y[o] = x[o] + acc[v] * rc;
        }
    }
}

// ---------------------------------------------------------------------------
extern "C" void kernel_launch(void* const* d_in, const int* in_sizes, int n_in,
                              void* d_out, int out_size, void* d_ws, size_t ws_size,
                              hipStream_t stream) {
    const float* X  = (const float*)d_in[0];   // agents_state [32,512,256]
    const float* Wq = (const float*)d_in[1];
    const float* bq = (const float*)d_in[2];
    const float* Wk = (const float*)d_in[3];
    const float* bk = (const float*)d_in[4];
    const float* Wv = (const float*)d_in[5];
    const float* bv = (const float*)d_in[6];
    float* out = (float*)d_out;

    char* ws = (char*)d_ws;
    float*          x0  = (float*)ws;                                       // 16 MB
    float*          x1  = (float*)(ws + (size_t)16 * 1024 * 1024);          // 16 MB
    unsigned short* Kb  = (unsigned short*)(ws + (size_t)32 * 1024 * 1024); //  8 MB
    unsigned short* Vt  = (unsigned short*)(ws + (size_t)40 * 1024 * 1024); //  8 MB
    unsigned short* Xb  = (unsigned short*)(ws + (size_t)48 * 1024 * 1024); //  8 MB
    unsigned short* Wb  = (unsigned short*)(ws + (size_t)56 * 1024 * 1024); // 384 KB

    cvt_x_bf16<<<4096, 256, 0, stream>>>(X, Xb);        // 4 elems/thread
    cvt_w_bf16<<<256, 256, 0, stream>>>(Wq, Wk, Wv, Wb);

    // QKV projections: 512 blocks x 8 waves x (16x64 tile), y = {Q,K,V}
    dim3 gq(512, 3, 1);
    qkv_wmma<<<gq, 256, 0, stream>>>(Xb, Wb, bq, bk, bv, x0, Kb, Vt);

    const size_t lds_bytes = (size_t)(2 * 16 * SSTR) * 4
                           + (size_t)(2 * 16 * PSTR) * 2
                           + 2 * 16 * 4;                 // ~97.2 KB
    attn_layer<<<4096, 64, lds_bytes, stream>>>(x0, Kb, Vt, x1);
    attn_layer<<<4096, 64, lds_bytes, stream>>>(x1, Kb, Vt, x0);
    attn_layer<<<4096, 64, lds_bytes, stream>>>(x0, Kb, Vt, x1);
    attn_layer<<<4096, 64, lds_bytes, stream>>>(x1, Kb, Vt, out);
}
